// CostEstimationNet_33011118637623
// MI455X (gfx1250) — matrix operand order repaired and checked by reading, not compiled, not run on hardware
//
#include <hip/hip_runtime.h>
#include <cstdint>

typedef __attribute__((ext_vector_type(16))) _Float16 v16h;
typedef __attribute__((ext_vector_type(8)))  float    v8f;

constexpr int B_  = 4096;   // batch
constexpr int T_  = 200;    // seq len
constexpr int IN_ = 79;     // input size
constexpr int H_  = 10;     // hidden
constexpr int G_  = 40;     // 4*H gates
constexpr int M_  = B_ * T_;

__device__ __forceinline__ float softplus_f(float x) { return log1pf(expf(x)); }
__device__ __forceinline__ float sigmoid_f(float x)  { return 1.f / (1.f + expf(-x)); }

// ---------------------------------------------------------------------------
// Kernel 1: materialize Wih = mu + softplus(rho)*eps, convert to f16, and
// store pre-swizzled into WMMA B-fragment layout:
//   flat = ((kc*3 + nt)*32 + lane)*16 + e
//   lane <16 : col = nt*16 + lane,      K = kc*32 + e        (e = 0..15)
//   lane>=16 : col = nt*16 + (lane-16), K = kc*32 + 16 + e
// K padded 79->96, N padded 40->48 with zeros.
// ---------------------------------------------------------------------------
__global__ void bayes_prep_wihB(const float* __restrict__ mu,
                                const float* __restrict__ rho,
                                const float* __restrict__ eps,
                                _Float16* __restrict__ wihB) {
  for (int f = threadIdx.x; f < 3 * 3 * 32 * 16; f += blockDim.x) {
    int e    = f & 15;
    int lane = (f >> 4) & 31;
    int i9   = f >> 9;          // kc*3 + nt, 0..8
    int kc   = i9 / 3;
    int nt   = i9 - kc * 3;
    int k = kc * 32 + ((lane < 16) ? e : 16 + e);
    int n = nt * 16 + (lane & 15);
    float v = 0.f;
    if (k < IN_ && n < G_) {
      int idx = k * G_ + n;
      v = mu[idx] + softplus_f(rho[idx]) * eps[idx];
    }
    wihB[f] = (_Float16)v;
  }
}

// ---------------------------------------------------------------------------
// Kernel 2: xg[t][b][g] = x[b][t][:] @ Wih + 0   (bias added in recurrence)
// WMMA f16 GEMM: per wave one 16(M)x48(N) tile, K loop 3 x 32.
// ---------------------------------------------------------------------------
__global__ void __launch_bounds__(256)
bayes_xproj_wmma(const float* __restrict__ x,
                 const _Float16* __restrict__ wihB,
                 float* __restrict__ xg) {
  const int lane = threadIdx.x & 31;
  const int wave = threadIdx.x >> 5;
  const int m0   = (blockIdx.x * 8 + wave) * 16;

  // B fragments: 9 x v16h, contiguous 32B per lane (pre-swizzled by kernel 1)
  v16h bf[9];
#pragma unroll
  for (int i = 0; i < 9; ++i)
    bf[i] = *(const v16h*)(wihB + (i * 32 + lane) * 16);

  const v8f zero = {0.f, 0.f, 0.f, 0.f, 0.f, 0.f, 0.f, 0.f};
  v8f acc0 = zero, acc1 = zero, acc2 = zero;

  // A fragment source: lane<16 row m0+lane holds K {kb..kb+15}
  // with kb = kc*32; lane>=16 same row, kb offset +16 (disjoint halves).
  const int   row  = m0 + (lane & 15);
  const float* xrow = x + (size_t)row * IN_;
  const int   s0   = (lane < 16) ? 0 : 16;

#pragma unroll
  for (int kc = 0; kc < 3; ++kc) {
    const int kb = kc * 32 + s0;
    v16h a;
#pragma unroll
    for (int j = 0; j < 8; ++j) {
      int k0 = kb + j;
      int k1 = kb + 8 + j;
      a[j]     = (k0 < IN_) ? (_Float16)xrow[k0] : (_Float16)0.f;
      a[8 + j] = (k1 < IN_) ? (_Float16)xrow[k1] : (_Float16)0.f;
    }
    acc0 = __builtin_amdgcn_wmma_f32_16x16x32_f16(false, a, false, bf[kc * 3 + 0],
                                                  (short)0, acc0, false, false);
    acc1 = __builtin_amdgcn_wmma_f32_16x16x32_f16(false, a, false, bf[kc * 3 + 1],
                                                  (short)0, acc1, false, false);
    acc2 = __builtin_amdgcn_wmma_f32_16x16x32_f16(false, a, false, bf[kc * 3 + 2],
                                                  (short)0, acc2, false, false);
  }

  // C/D layout: VGPR v -> row m0 + v (lanes 0-15) / m0 + v + 8 (lanes 16-31),
  // col = nt*16 + (lane&15). Store only cols < 40, remapped to [t][B][40].
  v8f accs[3] = {acc0, acc1, acc2};
#pragma unroll
  for (int nt = 0; nt < 3; ++nt) {
    int col = nt * 16 + (lane & 15);
    if (col < G_) {
#pragma unroll
      for (int v = 0; v < 8; ++v) {
        int mr = m0 + v + ((lane < 16) ? 0 : 8);
        int b  = mr / T_;
        int t  = mr - b * T_;
        xg[((size_t)t * B_ + b) * G_ + col] = accs[nt][v];
      }
    }
  }
}

// ---------------------------------------------------------------------------
// Kernel 3: LSTM recurrence. 4 lanes per batch element (one per gate group),
// recurrent weight slice + h + c in registers, gate exchange via LDS
// (wave-aligned groups). Final linear head at the end.
// ---------------------------------------------------------------------------
__global__ void __launch_bounds__(256)
bayes_lstm_recur(const float* __restrict__ xg,
                 const float* __restrict__ whh_mu, const float* __restrict__ whh_rho,
                 const float* __restrict__ whh_eps,
                 const float* __restrict__ b_mu, const float* __restrict__ b_rho,
                 const float* __restrict__ b_eps,
                 const float* __restrict__ lin_w, const float* __restrict__ lin_b,
                 float* __restrict__ out) {
  __shared__ float Whh_s[H_ * G_];
  __shared__ float b_s[G_];
  __shared__ float gates_s[64][G_];

  const int tid = threadIdx.x;
  for (int i = tid; i < H_ * G_; i += blockDim.x)
    Whh_s[i] = whh_mu[i] + softplus_f(whh_rho[i]) * whh_eps[i];
  if (tid < G_) b_s[tid] = b_mu[tid] + softplus_f(b_rho[tid]) * b_eps[tid];
  __syncthreads();

  const int group   = tid >> 2;             // 0..63  (wave-aligned: 8 groups/wave)
  const int ls      = tid & 3;              // gate section: 0=i 1=f 2=g 3=o
  const int batch   = blockIdx.x * 64 + group;
  const int colbase = ls * 10;

  // This lane's 10 recurrent-weight columns, held in registers (100 VGPRs).
  float wrec[H_][10];
#pragma unroll
  for (int k = 0; k < H_; ++k)
#pragma unroll
    for (int j = 0; j < 10; ++j)
      wrec[k][j] = Whh_s[k * G_ + colbase + j];

  float breg[10];
#pragma unroll
  for (int j = 0; j < 10; ++j) breg[j] = b_s[colbase + j];

  float h[H_], c[H_];
#pragma unroll
  for (int j = 0; j < H_; ++j) { h[j] = 0.f; c[j] = 0.f; }

  for (int t = 0; t < T_; ++t) {
    const float* gp = xg + ((size_t)t * B_ + batch) * G_ + colbase;
    float gate[10];
#pragma unroll
    for (int j = 0; j < 10; ++j) gate[j] = gp[j] + breg[j];
#pragma unroll
    for (int k = 0; k < H_; ++k) {
      float hk = h[k];
#pragma unroll
      for (int j = 0; j < 10; ++j) gate[j] = fmaf(hk, wrec[k][j], gate[j]);
    }
    if (ls == 2) {
#pragma unroll
      for (int j = 0; j < 10; ++j) gate[j] = tanhf(gate[j]);
    } else {
#pragma unroll
      for (int j = 0; j < 10; ++j) gate[j] = sigmoid_f(gate[j]);
    }
#pragma unroll
    for (int j = 0; j < 10; ++j) gates_s[group][colbase + j] = gate[j];
    __syncthreads();
    // All 4 lanes redundantly update c/h so h stays in registers everywhere.
#pragma unroll
    for (int j = 0; j < H_; ++j) {
      float ig = gates_s[group][j];
      float fg = gates_s[group][10 + j];
      float gg = gates_s[group][20 + j];
      float og = gates_s[group][30 + j];
      c[j] = fg * c[j] + ig * gg;
      h[j] = og * tanhf(c[j]);
    }
    __syncthreads();
  }

  if (ls == 0) {
    float acc = lin_b[0];
#pragma unroll
    for (int j = 0; j < H_; ++j) acc = fmaf(h[j], lin_w[j], acc);
    out[batch] = acc;
  }
}

// ---------------------------------------------------------------------------
extern "C" void kernel_launch(void* const* d_in, const int* in_sizes, int n_in,
                              void* d_out, int out_size, void* d_ws, size_t ws_size,
                              hipStream_t stream) {
  (void)in_sizes; (void)n_in; (void)out_size; (void)ws_size;
  const float* x       = (const float*)d_in[0];
  const float* wih_mu  = (const float*)d_in[1];
  const float* wih_rho = (const float*)d_in[2];
  const float* wih_eps = (const float*)d_in[3];
  const float* whh_mu  = (const float*)d_in[4];
  const float* whh_rho = (const float*)d_in[5];
  const float* whh_eps = (const float*)d_in[6];
  const float* b_mu    = (const float*)d_in[7];
  const float* b_rho   = (const float*)d_in[8];
  const float* b_eps   = (const float*)d_in[9];
  const float* lin_w   = (const float*)d_in[10];
  const float* lin_b   = (const float*)d_in[11];
  float* out = (float*)d_out;

  _Float16* wihB = (_Float16*)d_ws;                       // 9216 B of fragments
  float*    xg   = (float*)((char*)d_ws + 65536);         // [T][B][40] f32, 131 MB

  bayes_prep_wihB<<<1, 256, 0, stream>>>(wih_mu, wih_rho, wih_eps, wihB);
  bayes_xproj_wmma<<<M_ / 128, 256, 0, stream>>>(x, wihB, xg);
  bayes_lstm_recur<<<B_ / 64, 256, 0, stream>>>(xg, whh_mu, whh_rho, whh_eps,
                                                b_mu, b_rho, b_eps,
                                                lin_w, lin_b, out);
}
